// SuperLinear_3642132267085
// MI455X (gfx1250) — compile-verified, matrix operand order checked by reading, not run on hardware
//
#include <hip/hip_runtime.h>

typedef __bf16 v16bf __attribute__((ext_vector_type(16)));
typedef float  v8f   __attribute__((ext_vector_type(8)));

#define IN_DIMS   128
#define OUT_DIMS  32
#define NPOS      2048
#define ROWS_PB   128     // rows (b,o positions) per block
#define RPW       16      // rows per wave (one 16x16 WMMA M-tile)
#define THREADS   256     // 8 wave32
#define HS        136     // LDS row stride in bf16 elems (128 + 8 pad)
#define LN_EPS    1e-5f
#define WPK_PER_D 8192    // bf16 elems per d in packed weights (128*32*2 fmts)

// ---------------------------------------------------------------------------
// Pre-pass: w1[m,h,d] (f32, d contiguous) -> packed bf16 hi/lo fragments
// layout: wpk[d][kc(4)][ht(2)][fmt(2)][lane(32)][j(16)]  (16 KB per d)
// Reads are fully coalesced along d (128B lines fully used, once).
// ---------------------------------------------------------------------------
__global__ __launch_bounds__(256)
void pack_weights(const float* __restrict__ w1, __bf16* __restrict__ wpk)
{
    const int dd = threadIdx.x & 31;              // lane -> consecutive d
    const int er = threadIdx.x >> 5;
    const int d  = blockIdx.y * 32 + dd;
    const int e0 = blockIdx.x * 512;

    for (int i = 0; i < 64; ++i) {
        const int e = e0 + i * 8 + er;            // e = m*32 + h
        const float wv = w1[(size_t)e * NPOS + d];
        const __bf16 hi = (__bf16)wv;
        const __bf16 lo = (__bf16)(wv - (float)hi);

        const int m = e >> 5, h = e & 31;
        const int kc = m >> 5, k = m & 31;
        const int ht = h >> 4, l16 = h & 15;
        const int half = (k >> 3) & 1;            // K-halves per ISA B layout
        const int base = k - 8 * half;            // in {0..7, 16..23}
        const int j = (base & 7) + (base >= 16 ? 8 : 0);

        const size_t off = (size_t)d * WPK_PER_D
                         + (size_t)((kc * 2 + ht) * 2) * 512
                         + (half * 16 + l16) * 16 + j;
        wpk[off]       = hi;                      // fmt 0
        wpk[off + 512] = lo;                      // fmt 1
    }
}

// ---------------------------------------------------------------------------
// Fused LayerNorm + grouped-linear GEMM.
// PACKED=1: B fragments loaded directly from wpk (no LDS for B, no barrier).
// PACKED=0: fallback, gather w1 into LDS (original path).
// ---------------------------------------------------------------------------
template <bool PACKED>
__global__ __launch_bounds__(THREADS)
void superlinear_ln_wmma(const float* __restrict__ x,
                         const float* __restrict__ w1,
                         const float* __restrict__ b1,
                         const float* __restrict__ T,
                         const float* __restrict__ gamma,
                         const float* __restrict__ beta,
                         const __bf16* __restrict__ wpk,
                         float* __restrict__ out)
{
    __shared__ __bf16 Hhi[ROWS_PB][HS];
    __shared__ __bf16 Hlo[ROWS_PB][HS];
    __shared__ __bf16 Bhi[PACKED ? 1 : OUT_DIMS][HS];
    __shared__ __bf16 Blo[PACKED ? 1 : OUT_DIMS][HS];

    const int d    = blockIdx.y;
    const int pb   = blockIdx.x;
    const int tid  = threadIdx.x;
    const int lane = tid & 31;
    const int wave = tid >> 5;
    const int l16  = lane & 15;
    const int half = lane >> 4;

    if constexpr (!PACKED) {
        // gather w1[:, :, d] -> bf16 hi/lo, transposed [h][m]
        #pragma unroll
        for (int i = 0; i < 16; ++i) {
            int e = tid + i * THREADS;            // e = m*32 + h
            int m = e >> 5;
            int h = e & 31;
            float wv = w1[(size_t)e * NPOS + d];
            __bf16 hi = (__bf16)wv;
            __bf16 lo = (__bf16)(wv - (float)hi);
            Bhi[h][m] = hi;
            Blo[h][m] = lo;
        }
    }

    const float4 g4  = ((const float4*)gamma)[lane];
    const float4 be4 = ((const float4*)beta)[lane];

    // ---------------- LayerNorm: 16 rows per wave ------------------------
    const int prow0 = pb * ROWS_PB + wave * RPW;
    #pragma unroll 4
    for (int j = 0; j < RPW; ++j) {
        const int p = prow0 + j;
        const float* xr = x + ((size_t)p * NPOS + d) * IN_DIMS;
        if (j + 2 < RPW) {
            const float* xn = x + ((size_t)(p + 2) * NPOS + d) * IN_DIMS;
            __builtin_prefetch(xn + lane * 4, 0, 0);   // global_prefetch_b8
        }
        float4 xv = ((const float4*)xr)[lane];         // coalesced 512B/row

        float s  = xv.x + xv.y + xv.z + xv.w;
        float s2 = xv.x * xv.x + xv.y * xv.y + xv.z * xv.z + xv.w * xv.w;
        #pragma unroll
        for (int off = 16; off > 0; off >>= 1) {
            s  += __shfl_xor(s,  off, 32);
            s2 += __shfl_xor(s2, off, 32);
        }
        const float mu   = s * (1.0f / IN_DIMS);
        const float var  = s2 * (1.0f / IN_DIMS) - mu * mu;
        const float rstd = rsqrtf(var + LN_EPS);

        float hv[4];
        hv[0] = (xv.x - mu) * rstd * g4.x + be4.x;
        hv[1] = (xv.y - mu) * rstd * g4.y + be4.y;
        hv[2] = (xv.z - mu) * rstd * g4.z + be4.z;
        hv[3] = (xv.w - mu) * rstd * g4.w + be4.w;

        const int r = wave * RPW + j;
        #pragma unroll
        for (int k = 0; k < 4; ++k) {
            __bf16 hi = (__bf16)hv[k];
            __bf16 lo = (__bf16)(hv[k] - (float)hi);
            Hhi[r][lane * 4 + k] = hi;
            Hlo[r][lane * 4 + k] = lo;
        }
    }
    if constexpr (!PACKED) __syncthreads();   // packed: H rows are wave-local

    // ---------------- GEMM: 16x32 tile per wave, K=128 --------------------
    v8f acc0 = {};
    v8f acc1 = {};
    const int rowA = wave * RPW + l16;
    const __bf16* wd = PACKED ? (wpk + (size_t)d * WPK_PER_D) : nullptr;

    #pragma unroll
    for (int kc = 0; kc < 4; ++kc) {
        const int kbase = kc * 32 + half * 8;

        v16bf ahi, alo;
        {
            const __bf16* ph = &Hhi[rowA][kbase];
            const __bf16* pl = &Hlo[rowA][kbase];
            #pragma unroll
            for (int j = 0; j < 8; ++j) {
                ahi[j]     = ph[j];
                ahi[j + 8] = ph[j + 16];
                alo[j]     = pl[j];
                alo[j + 8] = pl[j + 16];
            }
        }

        #pragma unroll
        for (int ht = 0; ht < 2; ++ht) {
            v16bf bhi, blo;
            if constexpr (PACKED) {
                const __bf16* fb = wd + (size_t)((kc * 2 + ht) * 2) * 512
                                      + lane * 16;
                bhi = *(const v16bf*)(fb);         // 32B/lane, coalesced
                blo = *(const v16bf*)(fb + 512);
            } else {
                const __bf16* qh = &Bhi[ht * 16 + l16][kbase];
                const __bf16* ql = &Blo[ht * 16 + l16][kbase];
                #pragma unroll
                for (int j = 0; j < 8; ++j) {
                    bhi[j]     = qh[j];
                    bhi[j + 8] = qh[j + 16];
                    blo[j]     = ql[j];
                    blo[j + 8] = ql[j + 16];
                }
            }
            v8f acc = ht ? acc1 : acc0;
            // fp32-accurate via bf16 split: ah*bh + ah*bl + al*bh
            acc = __builtin_amdgcn_wmma_f32_16x16x32_bf16(
                      false, ahi, false, bhi, (short)0, acc, false, false);
            acc = __builtin_amdgcn_wmma_f32_16x16x32_bf16(
                      false, ahi, false, blo, (short)0, acc, false, false);
            acc = __builtin_amdgcn_wmma_f32_16x16x32_bf16(
                      false, alo, false, bhi, (short)0, acc, false, false);
            if (ht) acc1 = acc; else acc0 = acc;
        }
    }

    // ---------------- epilogue: bias, 1/T, store ---------------------------
    const float invT = 1.0f / T[0];
    #pragma unroll
    for (int ht = 0; ht < 2; ++ht) {
        const int n = ht * 16 + l16;
        const float bias = b1[d * OUT_DIMS + n];
        const v8f acc = ht ? acc1 : acc0;
        #pragma unroll
        for (int j = 0; j < 8; ++j) {
            const int p = prow0 + j + 8 * half;    // C: VGPR j -> M=j+8*half
            out[((size_t)p * NPOS + d) * OUT_DIMS + n] = (acc[j] + bias) * invT;
        }
    }
}

extern "C" void kernel_launch(void* const* d_in, const int* in_sizes, int n_in,
                              void* d_out, int out_size, void* d_ws, size_t ws_size,
                              hipStream_t stream) {
    const float* x     = (const float*)d_in[0];
    const float* w1    = (const float*)d_in[1];
    const float* b1    = (const float*)d_in[2];
    const float* T     = (const float*)d_in[3];
    const float* gamma = (const float*)d_in[4];
    const float* beta  = (const float*)d_in[5];
    float* out = (float*)d_out;

    dim3 grid(512 / ROWS_PB, NPOS);   // 4 row-blocks x 2048 positions
    dim3 block(THREADS);

    const size_t need = (size_t)NPOS * WPK_PER_D * sizeof(__bf16);  // 32 MB
    if (ws_size >= need) {
        __bf16* wpk = (__bf16*)d_ws;
        pack_weights<<<dim3(8, 64), 256, 0, stream>>>(w1, wpk);
        superlinear_ln_wmma<true><<<grid, block, 0, stream>>>(
            x, w1, b1, T, gamma, beta, wpk, out);
    } else {
        superlinear_ln_wmma<false><<<grid, block, 0, stream>>>(
            x, w1, b1, T, gamma, beta, nullptr, out);
    }
}
